// sLSTM_18906446037401
// MI455X (gfx1250) — matrix operand order repaired and checked by reading, not compiled, not run on hardware
//
#include <hip/hip_runtime.h>

// ---------------------------------------------------------------------------
// Bidirectional(-style) LSTM for MI455X (gfx1250, wave32, WMMA).
//
//  K1: bulk GEMM  Xproj[d,t,b,4H] = x[b,t,:] @ Wx[d] + bias[d]  with
//      v_wmma_f32_16x16x32_bf16, stored in C-fragment layout as bf16 (256MB).
//  K2: recurrent phase. Key insight: the recurrence for (dir d, 16-row batch
//      tile ib) depends only on h of that same (d, ib) group. So one
//      512-thread workgroup (16 waves = all jh tiles of one (d,ib)) owns an
//      entire independent recurrence: h_prev lives in a double-buffered,
//      pre-swizzled A-fragment LDS buffer (16KB), steps are separated by a
//      single hardware s_barrier, and cell state c never leaves VGPRs.
//      No grid-wide sync, no global h traffic on the serial critical path.
//      Per wave per step: 32x WMMA (4 gate tiles x K=256).
//  K0: one-time weight swizzle f32->bf16 into exact B-fragment lane layout.
//
// Workspace: 2MB (weights) + 256MB (Xproj frags) ~= 258 MB.
// ---------------------------------------------------------------------------

typedef __bf16 bf16_t;
typedef __attribute__((ext_vector_type(16))) __bf16 v16bf;
typedef __attribute__((ext_vector_type(8)))  float  v8f;

#define B_  128
#define T_  512
#define I_  256
#define H_  256

union FragAB { v16bf v; uint4 q[2]; bf16_t e[16]; };
union FragC  { v8f   v; float e[8]; };
union Half8  { uint4 q; bf16_t e[8]; };

__device__ __forceinline__ float sigm(float x)  { return 1.0f / (1.0f + __expf(-x)); }
__device__ __forceinline__ float tanh_(float x) { return 1.0f - 2.0f / (__expf(2.0f * x) + 1.0f); }

// ---------------------------------------------------------------------------
// K0: swizzle W (512x1024 f32, rows [Wx;Wh]) into bf16 B-fragment layout:
//   frag[(d*64+nb)*8+kb][lane][slot], lane = bit3(km32)*16 + n%16, slot = (km32&7) + 8*(km32>=16)
// (ISA 7.12.2: B-matrix 16-bit, lanes 0-15 hold K 0-7,16-23, lanes 16-31 K 8-15,24-31.)
// ---------------------------------------------------------------------------
__global__ void prep_kernel(const float* __restrict__ Wf, const float* __restrict__ Wb,
                            bf16_t* __restrict__ wxfrag, bf16_t* __restrict__ whfrag) {
    unsigned tid = blockIdx.x * blockDim.x + threadIdx.x;
    if (tid < 2u * 512u * 1024u) {
        unsigned d    = tid >> 19;
        unsigned rest = tid & ((1u << 19) - 1u);
        unsigned k    = rest >> 10;     // 0..511 (0..255 = Wx rows, 256..511 = Wh rows)
        unsigned n    = rest & 1023u;   // column
        const float* W = d ? Wb : Wf;
        float val = W[(size_t)k * 1024 + n];
        unsigned kk   = k & 255u;
        unsigned kb   = kk >> 5;
        unsigned km   = kk & 31u;
        unsigned nb   = n >> 4;
        unsigned lane = (((km >> 3) & 1u) << 4) | (n & 15u);
        unsigned slot = (km & 7u) + 8u * (km >> 4);
        bf16_t* dst = (k < 256u) ? wxfrag : whfrag;
        dst[(((size_t)d * 64 + nb) * 8 + kb) * 512 + lane * 16 + slot] = (bf16_t)val;
    }
}

// ---------------------------------------------------------------------------
// K1: Xproj GEMM. One wave per 16x16 output tile (d, t, ib, nb).
// A-frag of x per the 16-bit A layout: lane half h holds K runs
// [8h..8h+7] and [16+8h..16+8h+7] -> two contiguous 8-float segments.
// ---------------------------------------------------------------------------
__global__ void __launch_bounds__(256) xproj_kernel(
        const float* __restrict__ x, const float* __restrict__ bfv,
        const float* __restrict__ bbv, const bf16_t* __restrict__ wxfrag,
        bf16_t* __restrict__ xfrag) {
    unsigned lane = threadIdx.x & 31u;
    unsigned idx  = blockIdx.x * 8u + (threadIdx.x >> 5);  // 0..524287
    unsigned nb = idx & 63u;
    unsigned ib = (idx >> 6) & 7u;
    unsigned t  = (idx >> 9) & 511u;
    unsigned d  = (idx >> 18) & 1u;

    unsigned nl   = lane & 15u;
    unsigned half = lane >> 4;

    const float* bias = d ? bbv : bfv;
    FragC acc;
    float bv = bias[nb * 16 + nl];
#pragma unroll
    for (int r = 0; r < 8; r++) acc.e[r] = bv;

    unsigned brow = ib * 16 + nl;
    const float*  xrow = x + ((size_t)brow * T_ + t) * I_;
    const bf16_t* wb   = wxfrag + (((size_t)d * 64 + nb) * 8) * 512 + lane * 16;

#pragma unroll
    for (int kb = 0; kb < 8; kb++) {
        FragAB A, Bm;
        unsigned k0 = kb * 32 + 8 * half;
        const float4* p0 = (const float4*)(xrow + k0);
        const float4* p1 = (const float4*)(xrow + k0 + 16);
        float4 f0 = p0[0], f1 = p0[1], f2 = p1[0], f3 = p1[1];
        A.e[0]  = (bf16_t)f0.x; A.e[1]  = (bf16_t)f0.y; A.e[2]  = (bf16_t)f0.z; A.e[3]  = (bf16_t)f0.w;
        A.e[4]  = (bf16_t)f1.x; A.e[5]  = (bf16_t)f1.y; A.e[6]  = (bf16_t)f1.z; A.e[7]  = (bf16_t)f1.w;
        A.e[8]  = (bf16_t)f2.x; A.e[9]  = (bf16_t)f2.y; A.e[10] = (bf16_t)f2.z; A.e[11] = (bf16_t)f2.w;
        A.e[12] = (bf16_t)f3.x; A.e[13] = (bf16_t)f3.y; A.e[14] = (bf16_t)f3.z; A.e[15] = (bf16_t)f3.w;
        const uint4* wq = (const uint4*)(wb + (size_t)kb * 512);
        Bm.q[0] = wq[0]; Bm.q[1] = wq[1];
        acc.v = __builtin_amdgcn_wmma_f32_16x16x32_bf16(
            false, A.v, false, Bm.v, (short)0, acc.v, false, false);
    }

    Half8 st;
#pragma unroll
    for (int r = 0; r < 8; r++) st.e[r] = (bf16_t)acc.e[r];
    size_t fragidx = (((size_t)d * 512 + t) * 8 + ib) * 64 + nb;
    *((uint4*)(xfrag + fragidx * 256 + lane * 8)) = st.q;
}

// ---------------------------------------------------------------------------
// K2: recurrent kernel. Grid = 16 WGs (2 dirs x 8 b-tiles), 512 threads
// (16 waves, one per 16-col h-tile jh). Fully independent per WG.
// h double-buffered in LDS in A-fragment layout; c-state stays in VGPRs.
// ---------------------------------------------------------------------------
__global__ void __launch_bounds__(512) lstm_rec_kernel(
        const bf16_t* __restrict__ whfrag, const bf16_t* __restrict__ xfrag,
        float* __restrict__ out, float* __restrict__ hT, float* __restrict__ cT) {
    __shared__ bf16_t hlds[2][8][512];   // [parity][kb][lane*16+slot] = 16KB

    unsigned lane = threadIdx.x & 31u;
    unsigned jh   = threadIdx.x >> 5;    // 0..15
    unsigned d    = blockIdx.x >> 3;
    unsigned ib   = blockIdx.x & 7u;
    unsigned nl   = lane & 15u;

    // Zero parity-0 h buffer (t=0 reads zeros).
    {
        unsigned* z = (unsigned*)&hlds[0][0][0];   // 2048 dwords
        for (unsigned i = threadIdx.x; i < 2048u; i += 512u) z[i] = 0u;
    }
    __syncthreads();

    float c[8];
#pragma unroll
    for (int r = 0; r < 8; r++) c[r] = 0.0f;

    const bf16_t* whb = whfrag + ((size_t)d * 64 * 8) * 512 + lane * 16;
    const bf16_t* xfb = xfrag + ((size_t)d * 512 * 8 * 64) * 256 + lane * 8;

    // Pre-swizzled h write target (A-fragment layout, ISA 16-bit A table):
    // km32 = (jh&1)*16 + nl  ->  half = bit3(nl), slot = (nl&7) + 8*(jh&1)
    unsigned halfT = (nl >> 3) & 1u;
    unsigned slotT = (nl & 7u) + 8u * (jh & 1u);
    unsigned kbT   = jh >> 1;
    unsigned mbase = (lane >> 4) * 8u;   // C/D rows: M = r + 8*(lane>=16)
    unsigned col   = jh * 16 + nl;

    for (unsigned t = 0; t < 512u; ++t) {
        unsigned par = t & 1u;

        // A-fragments of h_prev straight from LDS (2x ds b128 per kb).
        FragAB A[8];
#pragma unroll
        for (int kb = 0; kb < 8; kb++) {
            const uint4* p = (const uint4*)&hlds[par][kb][lane * 16];
            A[kb].q[0] = p[0]; A[kb].q[1] = p[1];
        }

        // 4 gate tiles: C init from Xproj fragment, then 8 WMMAs over K=256.
        FragC acc[4];
#pragma unroll
        for (int g = 0; g < 4; ++g) {
            unsigned nb = g * 16 + jh;
            Half8 xl;
            xl.q = *((const uint4*)(xfb + (((size_t)t * 8 + ib) * 64 + nb) * 256));
#pragma unroll
            for (int r = 0; r < 8; r++) acc[g].e[r] = (float)xl.e[r];
            const bf16_t* wg = whb + (size_t)nb * 8 * 512;
#pragma unroll
            for (int kb = 0; kb < 8; kb++) {
                FragAB Bm;
                const uint4* wq = (const uint4*)(wg + (size_t)kb * 512);
                Bm.q[0] = wq[0]; Bm.q[1] = wq[1];
                acc[g].v = __builtin_amdgcn_wmma_f32_16x16x32_bf16(
                    false, A[kb].v, false, Bm.v, (short)0, acc[g].v, false, false);
            }
        }

        // LSTM elementwise; h -> global out + next-parity LDS A-frag buffer.
        bf16_t* hw = &hlds[par ^ 1u][kbT][0];
#pragma unroll
        for (int r = 0; r < 8; r++) {
            float iv = sigm(acc[0].e[r]);
            float fv = sigm(acc[1].e[r]);
            float gv = tanh_(acc[2].e[r]);
            float ov = sigm(acc[3].e[r]);
            c[r] = fv * c[r] + iv * gv;
            float hv = ov * tanh_(c[r]);
            unsigned mrow = mbase + r;
            unsigned brow = ib * 16 + mrow;
            out[((size_t)brow * T_ + t) * (2 * H_) + (size_t)d * H_ + col] = hv;
            hw[(halfT * 16 + mrow) * 16 + slotT] = (bf16_t)hv;
            if (t == 511u) {
                hT[((size_t)d * B_ + brow) * H_ + col] = hv;
                cT[((size_t)d * B_ + brow) * H_ + col] = c[r];
            }
        }

        // Hardware workgroup barrier: next step may read parity^1.
        __syncthreads();
    }
}

// ---------------------------------------------------------------------------
extern "C" void kernel_launch(void* const* d_in, const int* in_sizes, int n_in,
                              void* d_out, int out_size, void* d_ws, size_t ws_size,
                              hipStream_t stream) {
    (void)in_sizes; (void)n_in; (void)out_size; (void)ws_size;
    const float* x  = (const float*)d_in[0];
    const float* Wf = (const float*)d_in[1];
    const float* bf = (const float*)d_in[2];
    const float* Wb = (const float*)d_in[3];
    const float* bb = (const float*)d_in[4];

    float* out = (float*)d_out;                      // [B, T, 2H]
    float* hT  = out + (size_t)B_ * T_ * 2 * H_;     // [2, B, H]
    float* cT  = hT + (size_t)2 * B_ * H_;           // [2, B, H]

    char* ws = (char*)d_ws;
    bf16_t* wxfrag = (bf16_t*)(ws);                      // 1MB
    bf16_t* whfrag = (bf16_t*)(ws + ((size_t)1 << 20));  // 1MB
    bf16_t* xfrag  = (bf16_t*)(ws + ((size_t)2 << 20));  // 256MB

    prep_kernel<<<4096, 256, 0, stream>>>(Wf, Wb, wxfrag, whfrag);
    xproj_kernel<<<65536, 256, 0, stream>>>(x, bf, bb, wxfrag, xfrag);
    lstm_rec_kernel<<<16, 512, 0, stream>>>(whfrag, xfrag, out, hT, cT);
}
